// Generator_11201274708113
// MI455X (gfx1250) — matrix operand order repaired and checked by reading, not compiled, not run on hardware
//
#include <hip/hip_runtime.h>

// ---------------------------------------------------------------------------
// gfx1250 StyleGAN-like generator forward pass.
// Activations stored f16 (bandwidth-bound @ 23.3 TB/s HBM); math in f32.
// Convs: implicit GEMM on v_wmma_f32_16x16x32_f16, multi-wave workgroups
// sharing an LDS-staged, double-buffered A (im2col) tile, filled with
// GLOBAL_LOAD_ASYNC_TO_LDS_B128 when the toolchain exposes the builtin.
// ---------------------------------------------------------------------------

typedef __attribute__((ext_vector_type(16))) _Float16 v16h;
typedef __attribute__((ext_vector_type(8)))  _Float16 v8h;
typedef __attribute__((ext_vector_type(8)))  float    v8f;

union AFrag { v16h v; v8h h8[2]; _Float16 h[16]; };
union CFrag { v8f  v; float f[8]; };

#define B8 8   // batch

#if defined(__has_builtin)
#if __has_builtin(__builtin_amdgcn_global_load_async_to_lds_b128)
#define HAVE_ASYNC_LDS 1
#endif
#endif
#ifndef HAVE_ASYNC_LDS
#define HAVE_ASYNC_LDS 0
#endif

#if HAVE_ASYNC_LDS
// Builtin signature (from compiler diagnostic): param 0 is
// 'int __vector(4) __device__ *' (global AS), param 1 is the LDS-side pointer.
typedef int v4i __attribute__((vector_size(16)));
typedef __attribute__((address_space(1))) v4i gas_v4i;
typedef __attribute__((address_space(3))) v4i las_v4i;
#endif

__device__ __forceinline__ void wait_async_lds() {
#if HAVE_ASYNC_LDS
#if __has_builtin(__builtin_amdgcn_s_wait_asynccnt)
  __builtin_amdgcn_s_wait_asynccnt(0);
#else
  asm volatile("s_wait_asynccnt 0x0" ::: "memory");
#endif
#endif
}

// ---------------- elementwise / small kernels ------------------------------

__global__ void g_zero(float* __restrict__ p, int n) {
  int i = blockIdx.x * blockDim.x + threadIdx.x;
  if (i < n) p[i] = 0.f;
}

// S[b, 0:2d] = styles[b,:] @ W + bias      (W is [512, 2d] row-major)
__global__ void g_style(const float* __restrict__ styles, const float* __restrict__ W,
                        const float* __restrict__ bias, float* __restrict__ S, int twoD) {
  int tid = blockIdx.x * blockDim.x + threadIdx.x;
  if (tid >= B8 * twoD) return;
  int c = tid % twoD, b = tid / twoD;
  const float* st = styles + b * 512;
  float acc = bias[c];
  for (int k = 0; k < 512; ++k) acc += st[k] * W[(size_t)k * twoD + c];
  S[tid] = acc;
}

// block0 input: lrelu(1 + noise * na1)  -> f16
__global__ void g_fill0(_Float16* __restrict__ x, const float* __restrict__ nz,
                        const float* __restrict__ na, int HW, int C) {
  long i = (long)blockIdx.x * blockDim.x + threadIdx.x;
  long n = (long)B8 * HW * C;
  if (i >= n) return;
  int c = (int)(i % C);
  int p = (int)((i / C) % HW);
  float v = 1.f + nz[p] * na[c];
  x[i] = (_Float16)(v >= 0.f ? v : 0.2f * v);
}

// x = lrelu(x + noise * na)
__global__ void g_noise_lrelu(_Float16* __restrict__ x, const float* __restrict__ nz,
                              const float* __restrict__ na, int HW, int C) {
  long i = (long)blockIdx.x * blockDim.x + threadIdx.x;
  long n = (long)B8 * HW * C;
  if (i >= n) return;
  int c = (int)(i % C);
  int p = (int)((i / C) % HW);
  float v = (float)x[i] + nz[p] * na[c];
  x[i] = (_Float16)(v >= 0.f ? v : 0.2f * v);
}

// 2x nearest upsample (H,W are OUTPUT dims)
__global__ void g_up2(const _Float16* __restrict__ in, _Float16* __restrict__ out,
                      int H, int W, int C) {
  long i = (long)blockIdx.x * blockDim.x + threadIdx.x;
  long n = (long)B8 * H * W * C;
  if (i >= n) return;
  int c = (int)(i % C);
  long t = i / C;
  int x = (int)(t % W); t /= W;
  int y = (int)(t % H);
  int b = (int)(t / H);
  out[i] = in[(((long)b * (H >> 1) + (y >> 1)) * (W >> 1) + (x >> 1)) * C + c];
}

// depthwise 3x3 binomial blur, SAME, zero pad
__global__ void g_blur(const _Float16* __restrict__ in, _Float16* __restrict__ out,
                       int H, int W, int C) {
  long i = (long)blockIdx.x * blockDim.x + threadIdx.x;
  long n = (long)B8 * H * W * C;
  if (i >= n) return;
  int c = (int)(i % C);
  long t = i / C;
  int x = (int)(t % W); t /= W;
  int y = (int)(t % H);
  int b = (int)(t / H);
  const float k1[3] = {1.f, 2.f, 1.f};
  float s = 0.f;
  for (int dy = -1; dy <= 1; ++dy) {
    int yy = y + dy;
    if (yy < 0 || yy >= H) continue;
    for (int dx = -1; dx <= 1; ++dx) {
      int xx = x + dx;
      if (xx < 0 || xx >= W) continue;
      s += (float)in[(((long)b * H + yy) * W + xx) * C + c] * (k1[dy + 1] * k1[dx + 1]);
    }
  }
  out[i] = (_Float16)(s * (1.f / 16.f));
}

// instance norm partial sums: blockDim.x == C, grid = (B, chunks)
__global__ void g_stats(const _Float16* __restrict__ x, float* __restrict__ stats,
                        int HW, int C, int chunks) {
  int b = blockIdx.x, ch = blockIdx.y, c = threadIdx.x;
  int per = (HW + chunks - 1) / chunks;
  int p0 = ch * per;
  int p1 = p0 + per; if (p1 > HW) p1 = HW;
  float s = 0.f, s2 = 0.f;
  const _Float16* base = x + (long)b * HW * C + c;
  for (int p = p0; p < p1; ++p) {
    float v = (float)base[(long)p * C];
    s += v; s2 += v * v;
  }
  atomicAdd(&stats[(b * C + c) * 2 + 0], s);
  atomicAdd(&stats[(b * C + c) * 2 + 1], s2);
}

// x = S[b,c] + ((x - mean) * rsqrt(var + eps)) * S[b, C + c]
__global__ void g_norm_mod(_Float16* __restrict__ x, const float* __restrict__ stats,
                           const float* __restrict__ S, int HW, int C) {
  long i = (long)blockIdx.x * blockDim.x + threadIdx.x;
  long n = (long)B8 * HW * C;
  if (i >= n) return;
  int c = (int)(i % C);
  int b = (int)(i / ((long)HW * C));
  float inv = 1.f / (float)HW;
  float m = stats[(b * C + c) * 2 + 0] * inv;
  float v = stats[(b * C + c) * 2 + 1] * inv - m * m;
  float y = ((float)x[i] - m) * rsqrtf(v + 1e-8f);
  x[i] = (_Float16)(S[(long)b * 2 * C + c] + y * S[(long)b * 2 * C + C + c]);
}

// 1x1 conv 32->3 + bias (f16 in, f32 out)
__global__ void g_rgb(const _Float16* __restrict__ x, const float* __restrict__ w,
                      const float* __restrict__ bias, float* __restrict__ out, int N) {
  long tid = (long)blockIdx.x * blockDim.x + threadIdx.x;
  long tot = (long)N * 3;
  if (tid >= tot) return;
  int o = (int)(tid % 3);
  long pix = tid / 3;
  const _Float16* xp = x + pix * 32;
  float s = bias[o];
  for (int c = 0; c < 32; ++c) s += (float)xp[c] * w[c * 3 + o];
  out[tid] = s;
}

// ---------------- weight prep ----------------------------------------------

// StyleGAN transform_kernel: 3x3 -> 4x4 (pad + 4 shifted sums), spatially
// flipped so the deconv becomes a forward conv on the lhs-dilated input.
__global__ void g_wt4(const float* __restrict__ w, float* __restrict__ wt,
                      int Cin, int Cout) {
  int i = blockIdx.x * blockDim.x + threadIdx.x;
  int tot = 16 * Cin * Cout;
  if (i >= tot) return;
  int co = i % Cout;
  int r = i / Cout;
  int ci = r % Cin;
  int t = r / Cin;
  int ti = t >> 2, tj = t & 3;
  int fi = 3 - ti, fj = 3 - tj;   // flip for transpose conv
  float s = 0.f;
  for (int da = 0; da < 2; ++da)
    for (int db = 0; db < 2; ++db) {
      int wi = fi + da - 1, wj = fj + db - 1;
      if (wi >= 0 && wi < 3 && wj >= 0 && wj < 3)
        s += w[(((size_t)wi * 3 + wj) * Cin + ci) * Cout + co];
    }
  wt[i] = s;
}

// Pack f32 weights (k-major [K][Cout], k=(ky*ks+kx)*Cin+ci) into f16
// B-fragment order: pack[((ct*KT + kt)*32 + lane)*16 + h]
__global__ void g_pack(const float* __restrict__ w, _Float16* __restrict__ pack,
                       int K, int Cout) {
  int tid = blockIdx.x * blockDim.x + threadIdx.x;
  int KT = K >> 5;
  int tot = (Cout >> 4) * KT * 32;
  if (tid >= tot) return;
  int lane = tid & 31;
  int kt = (tid >> 5) % KT;
  int ct = (tid >> 5) / KT;
  int n = lane & 15;
  int kbase = kt * 32 + ((lane >> 4) << 4);
  _Float16* dst = pack + (size_t)tid * 16;
#pragma unroll
  for (int h = 0; h < 16; ++h)
    dst[h] = (_Float16)w[(size_t)(kbase + h) * Cout + ct * 16 + n];
}

// ---------------- implicit-GEMM conv on WMMA -------------------------------
// Workgroup = numWaves waves (64..256 threads). All waves share one 16-pixel
// M tile; wave w computes out-channel tile ct = blockIdx.y*numWaves + w.
// A (im2col, 16x32 f16) is staged in LDS (double-buffered) by threads 0..63,
// via async global->LDS copies when available. K = ks*ks*Cin (multiple of 32).
// dil=1: lhs-dilated input (stride-2 transpose conv), input is inH x inW.
__global__ __launch_bounds__(256)
void g_conv_wmma(const _Float16* __restrict__ in, const _Float16* __restrict__ wpack,
                 _Float16* __restrict__ out,
                 int H, int W, int wLog, int inH, int inW, int dil,
                 int Cin, int cinLog, int Cout, int ks, int pad, int numWaves) {
  __shared__ __align__(16) _Float16 ldsA[2][16 * 32];

  const int tid   = threadIdx.x;
  const int lane  = tid & 31;
  const int wave  = tid >> 5;
  const int tileM = blockIdx.x;
  const int ct    = blockIdx.y * numWaves + wave;
  const int b     = blockIdx.z;

  const int K  = ks * ks * Cin;
  const int KT = K >> 5;

  // ---- staging task coords (threads 0..63: 16 pixels x 4 groups of 8 k) ----
  const int sm = tid >> 2;                 // pixel index 0..15
  const int kg = tid & 3;                  // k-group 0..3
  const int sp = tileM * 16 + sm;
  const int spy = sp >> wLog;
  const int spx = sp & (W - 1);

  // ---- A-fragment coords ----
  const int m     = lane & 15;
  const int kbase = (lane < 16) ? 0 : 16;
  const int mHi   = (lane < 16) ? 0 : 8;

  auto stage = [&](int kt, int bufI) {
    if (tid < 64) {
      int k0 = kt * 32 + kg * 8;             // multiple of 8
      int ci = k0 & (Cin - 1);               // multiple of 8
      int s  = k0 >> cinLog;
      int ky = s / ks, kx = s - ky * ks;
      int iy, ix;
      bool ok;
      if (!dil) {
        iy = spy + ky - pad; ix = spx + kx - pad;
        ok = (iy >= 0) && (iy < inH) && (ix >= 0) && (ix < inW);
      } else {
        int sy = spy + ky - pad, sx = spx + kx - pad;
        ok = (sy >= 0) && (sx >= 0) && !(sy & 1) && !(sx & 1);
        iy = sy >> 1; ix = sx >> 1;
        ok = ok && (iy < inH) && (ix < inW);
      }
      _Float16* dst = &ldsA[bufI][sm * 32 + kg * 8];
      if (ok) {
        const _Float16* src = in + ((((size_t)b * inH + iy) * inW + ix) << cinLog) + ci;
#if HAVE_ASYNC_LDS
        __builtin_amdgcn_global_load_async_to_lds_b128((gas_v4i*)src, (las_v4i*)dst,
                                                       0, 0);
#else
        *(uint4*)dst = *(const uint4*)src;         // global_load_b128 + ds_store_b128
#endif
      } else {
        uint4 z; z.x = 0; z.y = 0; z.z = 0; z.w = 0;
        *(uint4*)dst = z;
      }
    }
  };

  CFrag acc;
#pragma unroll
  for (int r = 0; r < 8; ++r) acc.f[r] = 0.f;

  const _Float16* wbase = wpack + ((size_t)ct * KT * 32 + lane) * 16;

  stage(0, 0);
  wait_async_lds();
  __syncthreads();

  for (int kt = 0; kt < KT; ++kt) {
    const int cb = kt & 1;
    if (kt + 1 < KT) stage(kt + 1, cb ^ 1);

    AFrag a;
    a.h8[0] = *(const v8h*)&ldsA[cb][m * 32 + kbase];        // ds_load_b128
    a.h8[1] = *(const v8h*)&ldsA[cb][m * 32 + kbase + 8];    // ds_load_b128

    __builtin_prefetch(wbase + (size_t)(kt + 1) * 32 * 16, 0, 0);  // global_prefetch_b8
    v16h bf = *reinterpret_cast<const v16h*>(wbase + (size_t)kt * 32 * 16);

    acc.v = __builtin_amdgcn_wmma_f32_16x16x32_f16(
        /*neg_a=*/false, a.v, /*neg_b=*/false, bf,
        /*c_mod=*/(short)0, acc.v, /*reuse_a=*/false, /*reuse_b=*/false);

    wait_async_lds();
    __syncthreads();
  }

#pragma unroll
  for (int r = 0; r < 8; ++r) {
    int mm = r + mHi;
    int pp = tileM * 16 + mm;
    int yy = pp >> wLog, xx = pp & (W - 1);
    out[(((size_t)b * H + yy) * W + xx) * Cout + ct * 16 + (lane & 15)] =
        (_Float16)acc.f[r];
  }
}

// ---------------- host orchestration ---------------------------------------

static inline int ilog2i(int v) { int l = 0; while ((1 << l) < v) ++l; return l; }

extern "C" void kernel_launch(void* const* d_in, const int* in_sizes, int n_in,
                              void* d_out, int out_size, void* d_ws, size_t ws_size,
                              hipStream_t stream) {
  (void)in_sizes; (void)n_in; (void)out_size; (void)ws_size;

  static const int dims_[7] = {256, 256, 256, 256, 128, 64, 32};
  static const int res_[7]  = {4, 8, 16, 32, 64, 128, 256};

  // ---- inputs, setup_inputs() insertion order ----
  int idx = 0;
  const float* styles = (const float*)d_in[idx++];
  const float *upw[7] = {0}, *na1[7], *na2[7], *p1w[7], *p1b[7], *p2w[7], *p2b[7], *cw[7];
  for (int i = 0; i < 7; ++i) {
    if (i > 0) upw[i] = (const float*)d_in[idx++];
    na1[i] = (const float*)d_in[idx++];
    na2[i] = (const float*)d_in[idx++];
    p1w[i] = (const float*)d_in[idx++];
    p1b[i] = (const float*)d_in[idx++];
    p2w[i] = (const float*)d_in[idx++];
    p2b[i] = (const float*)d_in[idx++];
    cw[i]  = (const float*)d_in[idx++];
  }
  const float* rgbw = (const float*)d_in[idx++];
  const float* rgbb = (const float*)d_in[idx++];
  const float* noi[7];
  for (int i = 0; i < 7; ++i) noi[i] = (const float*)d_in[idx++];

  // ---- workspace layout (f16 activations: max 34 MiB per buffer) ----
  char* ws = (char*)d_ws;
  size_t off = 0;
  _Float16* bufA   = (_Float16*)(ws + off); off += (size_t)36 << 20;
  _Float16* bufB   = (_Float16*)(ws + off); off += (size_t)36 << 20;
  _Float16* wpack  = (_Float16*)(ws + off); off += (size_t)4 << 20;
  float*    wt4    = (float*)(ws + off);    off += (size_t)4 << 20;
  float*    styleS = (float*)(ws + off);    off += (size_t)1 << 20;
  float*    stats  = (float*)(ws + off);

  int sOff[7];
  { int o = 0; for (int i = 0; i < 7; ++i) { sOff[i] = o; o += 2 * B8 * 2 * dims_[i]; } }

  // ---- style projections (all blocks, both mods) ----
  for (int i = 0; i < 7; ++i) {
    int twoD = 2 * dims_[i], tot = B8 * twoD;
    g_style<<<(tot + 255) / 256, 256, 0, stream>>>(styles, p1w[i], p1b[i],
                                                   styleS + sOff[i], twoD);
    g_style<<<(tot + 255) / 256, 256, 0, stream>>>(styles, p2w[i], p2b[i],
                                                   styleS + sOff[i] + tot, twoD);
  }

  _Float16* cur = bufA;
  _Float16* alt = bufB;

  auto conv = [&](const float* wsrc, int H, int W, int inH, int inW, int dil,
                  int Cin, int Cout, int ks, int pad) {
    int K = ks * ks * Cin;
    int ptot = (Cout / 16) * (K / 32) * 32;
    g_pack<<<(ptot + 255) / 256, 256, 0, stream>>>(wsrc, wpack, K, Cout);
    int nw = (Cout / 16 < 8) ? Cout / 16 : 8;
    dim3 grid(H * W / 16, Cout / (16 * nw), B8);
    g_conv_wmma<<<grid, 32 * nw, 0, stream>>>(cur, wpack, alt, H, W, ilog2i(W),
                                              inH, inW, dil, Cin, ilog2i(Cin),
                                              Cout, ks, pad, nw);
    _Float16* t = cur; cur = alt; alt = t;
  };

  auto normMod = [&](int H, int W, int C, const float* S) {
    int HW = H * W, zc = B8 * C * 2;
    g_zero<<<(zc + 255) / 256, 256, 0, stream>>>(stats, zc);
    int chunks = HW / 16; if (chunks < 1) chunks = 1; if (chunks > 256) chunks = 256;
    g_stats<<<dim3(B8, chunks), C, 0, stream>>>(cur, stats, HW, C, chunks);
    long N = (long)B8 * HW * C;
    g_norm_mod<<<(N + 255) / 256, 256, 0, stream>>>(cur, stats, S, HW, C);
  };

  // ---- block 0 (4x4, 256ch, const input) ----
  {
    int H = 4, W = 4, C = 256, HW = 16;
    long N = (long)B8 * HW * C;
    g_fill0<<<(N + 255) / 256, 256, 0, stream>>>(cur, noi[0], na1[0], HW, C);
    normMod(H, W, C, styleS + sOff[0]);
    conv(cw[0], H, W, H, W, 0, C, C, 3, 1);
    g_noise_lrelu<<<(N + 255) / 256, 256, 0, stream>>>(cur, noi[0] + HW, na2[0], HW, C);
    normMod(H, W, C, styleS + sOff[0] + B8 * 2 * C);
  }

  // ---- blocks 1..6 ----
  for (int i = 1; i < 7; ++i) {
    int Cin = dims_[i - 1], C = dims_[i];
    int Hi = res_[i - 1], Ho = res_[i];
    int HW = Ho * Ho;
    long N = (long)B8 * HW * C;

    if (res_[i] < 128) {
      // repeat upsample + 3x3 conv
      long Nup = (long)B8 * HW * Cin;
      g_up2<<<(Nup + 255) / 256, 256, 0, stream>>>(cur, alt, Ho, Ho, Cin);
      { _Float16* t = cur; cur = alt; alt = t; }
      conv(upw[i], Ho, Ho, Ho, Ho, 0, Cin, C, 3, 1);
    } else {
      // deconv: build transformed 4x4 kernel, run lhs-dilated conv
      int wtTot = 16 * Cin * C;
      g_wt4<<<(wtTot + 255) / 256, 256, 0, stream>>>(upw[i], wt4, Cin, C);
      conv(wt4, Ho, Ho, Hi, Hi, 1, Cin, C, 4, 1);
    }

    g_blur<<<(N + 255) / 256, 256, 0, stream>>>(cur, alt, Ho, Ho, C);
    { _Float16* t = cur; cur = alt; alt = t; }

    g_noise_lrelu<<<(N + 255) / 256, 256, 0, stream>>>(cur, noi[i], na1[i], HW, C);
    normMod(Ho, Ho, C, styleS + sOff[i]);

    conv(cw[i], Ho, Ho, Ho, Ho, 0, C, C, 3, 1);

    g_noise_lrelu<<<(N + 255) / 256, 256, 0, stream>>>(cur, noi[i] + HW, na2[i], HW, C);
    normMod(Ho, Ho, C, styleS + sOff[i] + B8 * 2 * C);
  }

  // ---- to RGB ----
  {
    int N = B8 * 256 * 256;
    g_rgb<<<(((long)N * 3 + 255) / 256), 256, 0, stream>>>(cur, rgbw, rgbb,
                                                           (float*)d_out, N);
  }
}
